// ProposalTarget_71519795413735
// MI455X (gfx1250) — matrix operand order repaired and checked by reading, not compiled, not run on hardware
//
#include <hip/hip_runtime.h>
#include <math.h>

typedef __attribute__((ext_vector_type(16))) _Float16 v16h;
typedef __attribute__((ext_vector_type(8)))  float    v8f;

#define NP   100000
#define NG   512
#define NC   81
#define NCP  96          // NC padded to 6 WMMA tiles
#define NTOT (NP + NG)
#define RKEEP 128

// ---------------------------------------------------------------------------
// K1: per-proposal max-IoU + argmax over 512 gt boxes.
// gt boxes + areas staged in LDS (10 KB). One prop per lane, 393 blocks x 256.
// ---------------------------------------------------------------------------
__global__ __launch_bounds__(256) void k_iou(const float* __restrict__ props,
                                             const float* __restrict__ gt,
                                             float* __restrict__ maxov,
                                             int*   __restrict__ assign) {
  __shared__ float4 sgt[NG];
  __shared__ float  sarea[NG];
  const int tid = threadIdx.x;
  for (int g = tid; g < NG; g += 256) {
    float4 b;
    b.x = gt[g * 4 + 0]; b.y = gt[g * 4 + 1];
    b.z = gt[g * 4 + 2]; b.w = gt[g * 4 + 3];
    sgt[g]   = b;
    sarea[g] = (b.z - b.x + 1.f) * (b.w - b.y + 1.f);
  }
  __syncthreads();

  const int i = blockIdx.x * 256 + tid;
  if (i >= NTOT) return;

  const float* src = (i < NP) ? (props + (size_t)i * 4) : (gt + (size_t)(i - NP) * 4);
  const float ax1 = src[0], ay1 = src[1], ax2 = src[2], ay2 = src[3];
  const float areaA = (ax2 - ax1 + 1.f) * (ay2 - ay1 + 1.f);

  float best = -1.f;
  int   bi   = 0;
  for (int g = 0; g < NG; ++g) {
    const float4 b = sgt[g];
    float iw = fminf(ax2, b.z) - fmaxf(ax1, b.x) + 1.f;
    float ih = fminf(ay2, b.w) - fmaxf(ay1, b.y) + 1.f;
    iw = fmaxf(iw, 0.f);
    ih = fmaxf(ih, 0.f);
    const float inter = iw * ih;
    const float iou   = inter / (areaA + sarea[g] - inter);
    if (iou > best) { best = iou; bi = g; }  // strict '>' keeps first max (jnp.argmax)
  }
  maxov[i]  = best;
  assign[i] = bi;
}

// ---------------------------------------------------------------------------
// K1b: restage gt_labels as transposed, zero-padded f16: glT[96][512].
// Makes each WMMA B fragment one contiguous 32B per-lane load (no gather,
// no predication). {0,1} values are exact in f16.
// ---------------------------------------------------------------------------
__global__ __launch_bounds__(256) void k_glT(const float* __restrict__ gl,
                                             _Float16* __restrict__ glT) {
  const int idx = blockIdx.x * 256 + threadIdx.x;  // 96*512 = 49152
  if (idx >= NCP * NG) return;
  const int col = idx >> 9;      // 0..95
  const int k   = idx & (NG - 1);
  const float v = (col < NC) ? gl[(size_t)k * NC + col] : 0.f;
  glT[(size_t)col * NG + k] = (_Float16)v;
}

// ---------------------------------------------------------------------------
// K2: stable 3-way selection of the first 128 indices in (fg, bg, rest) order.
// Single block of 256 threads (8 wave32s); ballot-based prefix sums per
// 256-element chunk, running class counters carried sequentially.
// Also precomputes gt class ids (argmax of one-hot rows).
// ---------------------------------------------------------------------------
__global__ __launch_bounds__(256) void k_select(const float* __restrict__ maxov,
                                                const int*   __restrict__ assign,
                                                const float* __restrict__ gl,
                                                int* __restrict__ keep,
                                                int* __restrict__ keepFg,
                                                int* __restrict__ keepAssign,
                                                int* __restrict__ gtcls) {
  __shared__ int s_n0, s_n1;
  __shared__ int wc0[8], wc1[8], wc2[8];
  __shared__ int sc0, sc1, sc2;
  const int tid = threadIdx.x;
  if (tid == 0) { s_n0 = 0; s_n1 = 0; sc0 = 0; sc1 = 0; sc2 = 0; }
  __syncthreads();

  // Phase A: total fg / bg counts (bases for class-1 / class-2 positions).
  int c0 = 0, c1 = 0;
  for (int i = tid; i < NTOT; i += 256) {
    const float mv = maxov[i];
    if (mv >= 0.7f) c0++;
    else if (mv >= 0.1f && mv < 0.5f) c1++;
  }
  atomicAdd(&s_n0, c0);
  atomicAdd(&s_n1, c1);
  __syncthreads();
  const int n0 = s_n0, n1 = s_n1;

  const int lane = tid & 31;
  const int wave = tid >> 5;
  const unsigned long long lt = ((unsigned long long)1 << lane) - 1ull;
  const int nchunk = (NTOT + 255) / 256;

  for (int ch = 0; ch < nchunk; ++ch) {
    const int idx = ch * 256 + tid;
    int r = 3;  // out-of-range sentinel
    if (idx < NTOT) {
      const float mv = maxov[idx];
      r = (mv >= 0.7f) ? 0 : ((mv >= 0.1f && mv < 0.5f) ? 1 : 2);
    }
    const unsigned long long m0 = __ballot(r == 0);
    const unsigned long long m1 = __ballot(r == 1);
    const unsigned long long m2 = __ballot(r == 2);
    if (lane == 0) {
      wc0[wave] = __popcll(m0);
      wc1[wave] = __popcll(m1);
      wc2[wave] = __popcll(m2);
    }
    __syncthreads();
    int b0 = 0, b1 = 0, b2 = 0;
    for (int w = 0; w < wave; ++w) { b0 += wc0[w]; b1 += wc1[w]; b2 += wc2[w]; }
    int pos = -1;
    if      (r == 0) pos =           sc0 + b0 + __popcll(m0 & lt);
    else if (r == 1) pos = n0      + sc1 + b1 + __popcll(m1 & lt);
    else if (r == 2) pos = n0 + n1 + sc2 + b2 + __popcll(m2 & lt);
    if (pos >= 0 && pos < RKEEP) {
      keep[pos]   = idx;
      keepFg[pos] = (r == 0);
    }
    __syncthreads();
    if (tid == 0) {
      int t0 = 0, t1 = 0, t2 = 0;
      for (int w = 0; w < 8; ++w) { t0 += wc0[w]; t1 += wc1[w]; t2 += wc2[w]; }
      sc0 += t0; sc1 += t1; sc2 += t2;
    }
    __syncthreads();
  }

  if (tid < RKEEP) keepAssign[tid] = assign[keep[tid]];

  // gt class ids: first max of each one-hot row.
  for (int g = tid; g < NG; g += 256) {
    float best = gl[(size_t)g * NC];
    int   bi   = 0;
    for (int c = 1; c < NC; ++c) {
      const float v = gl[(size_t)g * NC + c];
      if (v > best) { best = v; bi = c; }
    }
    gtcls[g] = bi;
  }
}

// ---------------------------------------------------------------------------
// K3: rois gather, bbox_transform, bbox_targets zero + class-column scatter.
// ---------------------------------------------------------------------------
__global__ __launch_bounds__(128) void k_finalize(const float* __restrict__ props,
                                                  const float* __restrict__ gt,
                                                  const int* __restrict__ keep,
                                                  const int* __restrict__ keepFg,
                                                  const int* __restrict__ keepAssign,
                                                  const int* __restrict__ gtcls,
                                                  float* __restrict__ out) {
  const int p = threadIdx.x;
  if (p >= RKEEP) return;
  float* rois = out;                       // [128,4]
  float* bbt  = out + 512 + RKEEP * NC;    // [128, 4*81]

  const int i = keep[p];
  const float* src = (i < NP) ? (props + (size_t)i * 4) : (gt + (size_t)(i - NP) * 4);
  const float ex1 = src[0], ey1 = src[1], ex2 = src[2], ey2 = src[3];
  rois[p * 4 + 0] = ex1; rois[p * 4 + 1] = ey1;
  rois[p * 4 + 2] = ex2; rois[p * 4 + 3] = ey2;

  const int a = keepAssign[p];
  const float gx1 = gt[a * 4], gy1 = gt[a * 4 + 1], gx2 = gt[a * 4 + 2], gy2 = gt[a * 4 + 3];
  const float ew = ex2 - ex1 + 1.f, eh = ey2 - ey1 + 1.f;
  const float ecx = ex1 + 0.5f * ew, ecy = ey1 + 0.5f * eh;
  const float gw = gx2 - gx1 + 1.f, gh = gy2 - gy1 + 1.f;
  const float gcx = gx1 + 0.5f * gw, gcy = gy1 + 0.5f * gh;
  const float t0 = (gcx - ecx) / ew, t1 = (gcy - ecy) / eh;
  const float t2 = logf(gw / ew),    t3 = logf(gh / eh);

  const int cls = keepFg[p] ? gtcls[a] : 0;
  float* row = bbt + (size_t)p * 4 * NC;
  for (int c = 0; c < 4 * NC; ++c) row[c] = 0.f;
  if (cls > 0) {
    row[4 * cls + 0] = t0; row[4 * cls + 1] = t1;
    row[4 * cls + 2] = t2; row[4 * cls + 3] = t3;
  }
}

// ---------------------------------------------------------------------------
// K4: labels[128,81] = onehot(keepAssign)[128,512] @ gt_labels[512,81]
// via V_WMMA_F32_16X16X32_F16. One wave per 16x16 tile, grid (8,6),
// K looped 512/32 = 16 WMMA issues per wave.
//  - A built in registers: one v_cmp_eq + v_cndmask per element (fg folded
//    into a -1 sentinel so there is no divergence).
//  - B loaded from the transposed padded f16 buffer: one 32-byte contiguous
//    load per lane per k-step (glT[col][kbase..kbase+15]).
// bg rows get onehot(0) fixed up after accumulation (their A rows are zero).
// ---------------------------------------------------------------------------
__global__ __launch_bounds__(32) void k_labels_wmma(const _Float16* __restrict__ glT,
                                                    const int* __restrict__ keepFg,
                                                    const int* __restrict__ keepAssign,
                                                    float* __restrict__ labels) {
  const int lane = threadIdx.x;           // 0..31, whole wave active -> EXEC all 1s
  const int m0 = blockIdx.x * 16;         // 8 row tiles  (128 rows)
  const int n0 = blockIdx.y * 16;         // 6 col tiles  (81 cols, padded to 96)
  const int l15 = lane & 15;
  const int hi  = (lane >= 16);

  const int mrow   = m0 + l15;            // A: lanes 0-15 and 16-31 both carry M=0..15
  const int fgr    = keepFg[mrow];
  const int assEff = fgr ? keepAssign[mrow] : -1;   // sentinel: bg rows match nothing
  const int col    = n0 + l15;                      // B/D column (always < 96)

  // Per-lane K offsets of A-fragment elements (16x32 f16, this kernel's contiguous k order):
  // K = 16*(lane>=16) + j
  const int hi8  = hi ? 8 : 0;
  const _Float16* bsrc = glT + (size_t)col * NG;

  v8f c = {};
  for (int kk = 0; kk < NG / 32; ++kk) {
    v16h a;
    const int kbaseA = kk * 32 + (hi ? 16 : 0);
#pragma unroll
    for (int j = 0; j < 16; ++j) {
      const int v = j >> 1;
      const int K = kbaseA + j;
      a[j] = (assEff == K) ? (_Float16)1.0f : (_Float16)0.0f;
    }
    // B 32x16: lanes 0-15 hold K=0..15 (col per lane), lanes 16-31 K=16..31.
    const int kbaseB = kk * 32 + (hi ? 16 : 0);
    const v16h b = *(const v16h*)(bsrc + kbaseB);   // 32B contiguous, 32B aligned

    c = __builtin_amdgcn_wmma_f32_16x16x32_f16(false, a, false, b,
                                               (short)0, c, false, false);
  }

  if (col < NC) {
#pragma unroll
    for (int r = 0; r < 8; ++r) {
      const int row = m0 + r + hi8;            // C/D: VGPR r -> M = r + 8*(lane>=16)
      float v = c[r];
      if (col == 0 && !keepFg[row]) v = 1.0f;  // bg rows -> onehot(class 0)
      labels[(size_t)row * NC + col] = v;
    }
  }
}

// ---------------------------------------------------------------------------
extern "C" void kernel_launch(void* const* d_in, const int* in_sizes, int n_in,
                              void* d_out, int out_size, void* d_ws, size_t ws_size,
                              hipStream_t stream) {
  const float* props = (const float*)d_in[0];  // (1, 100000, 4) f32
  const float* gt    = (const float*)d_in[1];  // (1, 512, 4)    f32
  const float* gl    = (const float*)d_in[2];  // (1, 512, 81)   f32
  float* out = (float*)d_out;                  // rois(512) | labels(10368) | bbox(41472)

  char* ws = (char*)d_ws;
  size_t off = 0;
  auto take = [&](size_t bytes) -> char* {
    char* p = ws + off;
    off = (off + bytes + 255) & ~(size_t)255;
    return p;
  };
  float*     maxov      = (float*)    take(sizeof(float)    * NTOT);
  int*       assign     = (int*)      take(sizeof(int)      * NTOT);
  int*       keep       = (int*)      take(sizeof(int)      * RKEEP);
  int*       keepFg     = (int*)      take(sizeof(int)      * RKEEP);
  int*       keepAssign = (int*)      take(sizeof(int)      * RKEEP);
  int*       gtcls      = (int*)      take(sizeof(int)      * NG);
  _Float16*  glT        = (_Float16*) take(sizeof(_Float16) * NCP * NG);

  const int nblk = (NTOT + 255) / 256;
  k_iou<<<nblk, 256, 0, stream>>>(props, gt, maxov, assign);
  k_glT<<<(NCP * NG + 255) / 256, 256, 0, stream>>>(gl, glT);
  k_select<<<1, 256, 0, stream>>>(maxov, assign, gl, keep, keepFg, keepAssign, gtcls);
  k_finalize<<<1, 128, 0, stream>>>(props, gt, keep, keepFg, keepAssign, gtcls, out);
  k_labels_wmma<<<dim3(8, 6), 32, 0, stream>>>(glT, keepFg, keepAssign, out + 512);
}
